// HGTModule_5385888989305
// MI455X (gfx1250) — matrix operand order repaired and checked by reading, not compiled, not run on hardware
//
#include <hip/hip_runtime.h>
#include <hip/hip_bf16.h>
#include <math.h>

// ---------------------------------------------------------------------------
// HGT forward, item->user relation only (the user->item relation never
// reaches the returned output).  Dense stages: bf16 WMMA
// (v_wmma_f32_16x16x32_bf16) with weights staged in LDS in B-fragment order,
// coalesced b128 epilogue via per-wave LDS transpose.  Relation transforms
// folded into the GEMM weights.
// ---------------------------------------------------------------------------

typedef __attribute__((ext_vector_type(16))) __bf16 v16bf;
typedef __attribute__((ext_vector_type(8)))  float  v8f;

#define N_NODES 100000
#define EDGES   500000
#define HID     128
#define NH      4
#define HD      32

__device__ __forceinline__ float gelu_exact(float x) {
    return 0.5f * x * (1.0f + erff(x * 0.70710678118654752440f));
}
// monotonic float<->uint map for atomicMax on floats
__device__ __forceinline__ unsigned f2ord(float f) {
    unsigned u = __float_as_uint(f);
    return (u & 0x80000000u) ? ~u : (u | 0x80000000u);
}
__device__ __forceinline__ float ord2f(unsigned u) {
    unsigned v = (u & 0x80000000u) ? (u & 0x7FFFFFFFu) : ~u;
    return __uint_as_float(v);
}

// LDS weight store, swizzled so each lane's B-fragment is contiguous 32 bytes.
union WeightLds {
    __bf16 h[128 * 128];   // halfword view for staging writes
    v16bf  v[1024];        // fragment view: [kstep][ntile][lane]
};

// ---------------------------------------------------------------------------
// Fold per-head relation matrix R[h,32,32] into W[128,128] / b[128]:
//   Weff[k, h*32+e] = sum_d W[k, h*32+d] * R[h,d,e]
//   beff[h*32+e]    = sum_d b[h*32+d]    * R[h,d,e]
// ---------------------------------------------------------------------------
__global__ void fold_rel_kernel(const float* __restrict__ W,
                                const float* __restrict__ b,
                                const float* __restrict__ R,
                                float* __restrict__ Weff,
                                float* __restrict__ beff) {
    int idx = blockIdx.x * blockDim.x + threadIdx.x;  // 129*128 work items
    if (idx >= 129 * 128) return;
    int row = idx >> 7;
    int col = idx & 127;
    int h = col >> 5, e = col & 31;
    const float* Rh = R + h * 32 * 32;
    float s = 0.f;
    if (row < 128) {
        const float* wr = W + row * 128 + h * 32;
#pragma unroll
        for (int d = 0; d < 32; ++d) s += wr[d] * Rh[d * 32 + e];
        Weff[row * 128 + col] = s;
    } else {
        const float* br = b + h * 32;
#pragma unroll
        for (int d = 0; d < 32; ++d) s += br[d] * Rh[d * 32 + e];
        beff[col] = s;
    }
}

template <int PRE>
__device__ __forceinline__ void pack8(v16bf& a, int base, float4 p, float4 q) {
    float f[8] = {p.x, p.y, p.z, p.w, q.x, q.y, q.z, q.w};
#pragma unroll
    for (int j = 0; j < 8; ++j) {
        float v = (PRE == 1) ? gelu_exact(f[j]) : f[j];
        a[base + j] = (__bf16)v;
    }
}

// ---------------------------------------------------------------------------
// Y[N,128] = post( pre(X)[N,K] @ W[K,128] + bias ), optional skip-blend.
// PRE:  0 = identity, 1 = exact gelu (applied to X on load)
// POST: 0 = identity, 1 = relu
// BLEND: Y = sig(skip)*val + (1-sig(skip))*blendSrc
// Block = 256 threads (8 waves, wave32).  Wave w owns rows
// [blk*128 + w*16, +16) and all 128 output columns (8 WMMA tiles x K/32 steps).
// ---------------------------------------------------------------------------
template <int PRE, int POST, bool BLEND>
__global__ __launch_bounds__(256) void wmma_gemm_kernel(
        const float* __restrict__ X, const float* __restrict__ W,
        const float* __restrict__ bias, float* __restrict__ Y,
        int Nrows, int K,
        const float* __restrict__ blendSrc, const float* __restrict__ skipPtr) {
    __shared__ WeightLds Wsh;            // 32 KB, B-fragment-swizzled bf16
    __shared__ float stage[8 * 16 * 128];  // 64 KB, per-wave epilogue tile

    const int tid = threadIdx.x;
    // Cooperative linear read of W[k,n], swizzled write into fragment order:
    //   kstep s = k>>5, element e = (k&7)|((k>>1)&8), K-half kh = (k>>3)&1
    //   ntile t = n>>4, lane = kh*16 + (n&15)
    //   frag index = (s*8 + t),  halfword addr = frag*512 + lane*16 + e
    for (int i = tid; i < K * 128; i += 256) {
        int k = i >> 7, n = i & 127;
        int s = k >> 5, e = (k & 7) | ((k >> 1) & 8), kh = (k >> 3) & 1;
        int t = n >> 4, nl = n & 15;
        Wsh.h[((s * 8 + t) * 32 + (kh * 16 + nl)) * 16 + e] = (__bf16)W[i];
    }
    __syncthreads();

    const int wave = tid >> 5, lane = tid & 31;
    long rowBase = (long)blockIdx.x * 128 + wave * 16;
    if (rowBase >= Nrows) return;        // N is a multiple of 16

    const int m = lane & 15;             // A row / B col within tile
    const int khalf = lane >> 4;         // which K-half this lane holds
    const float* xrow = X + (rowBase + m) * (long)K;

    v8f acc[8];
#pragma unroll
    for (int t = 0; t < 8; ++t)
#pragma unroll
        for (int j = 0; j < 8; ++j) acc[t][j] = 0.f;

    const int ksteps = K >> 5;
    for (int s = 0; s < ksteps; ++s) {
        // A fragment: 16x32 bf16 (ISA 7.12.2): lanes 0-15 hold
        // K = s*32+{0..7,16..23}; lanes 16-31 hold K = s*32+{8..15,24..31}.
        int k0 = s * 32 + khalf * 8;
        const float4* xa = (const float4*)(xrow + k0);
        const float4* xb = (const float4*)(xrow + k0 + 16);
        float4 p0 = xa[0], p1 = xa[1];
        float4 q0 = xb[0], q1 = xb[1];
        v16bf a;
        pack8<PRE>(a, 0, p0, p1);
        pack8<PRE>(a, 8, q0, q1);
#pragma unroll
        for (int t = 0; t < 8; ++t) {
            v16bf bfrag = Wsh.v[(s * 8 + t) * 32 + lane];  // 2x ds_load_b128
            acc[t] = __builtin_amdgcn_wmma_f32_16x16x32_bf16(
                false, a, false, bfrag, (short)0, acc[t], false, false);
        }
    }

    // ---- epilogue: bias/act into per-wave LDS tile, then coalesced b128 ----
    float* st = stage + wave * (16 * 128);
#pragma unroll
    for (int t = 0; t < 8; ++t) {
        int n = t * 16 + m;
        float bn = bias[n];
#pragma unroll
        for (int j = 0; j < 8; ++j) {
            int r = j + 8 * khalf;       // D layout: VGPR j -> M = j + 8*khalf
            float v = acc[t][j] + bn;
            if (POST == 1) v = fmaxf(v, 0.f);
            st[r * 128 + n] = v;
        }
    }
    __asm volatile("s_wait_dscnt 0x0" ::: "memory");  // wave-local LDS RAW

    float ascale = 1.f, bscale = 0.f;
    if (BLEND) {
        float sk = skipPtr[0];
        ascale = 1.f / (1.f + __expf(-sk));
        bscale = 1.f - ascale;
    }
#pragma unroll
    for (int i = 0; i < 16; ++i) {       // lane covers cols [lane*4, +4) of row i
        float4 v = *(const float4*)&st[i * 128 + lane * 4];
        if (BLEND) {
            float4 bs = *(const float4*)&blendSrc[(rowBase + i) * HID + lane * 4];
            v.x = ascale * v.x + bscale * bs.x;
            v.y = ascale * v.y + bscale * bs.y;
            v.z = ascale * v.z + bscale * bs.z;
            v.w = ascale * v.w + bscale * bs.w;
        }
        *(float4*)&Y[(rowBase + i) * HID + lane * 4] = v;
    }
}

// ---------------------------------------------------------------------------
// Pass 1: per (edge, head) attention logit + per-(dst,head) running max.
// ---------------------------------------------------------------------------
__global__ void edge_logits_kernel(const float* __restrict__ k_rel,
                                   const float* __restrict__ q,
                                   const int* __restrict__ ei,
                                   const float* __restrict__ p_rel,
                                   float* __restrict__ logits,
                                   unsigned* __restrict__ node_max) {
    long idx = (long)blockIdx.x * blockDim.x + threadIdx.x;
    if (idx >= (long)EDGES * NH) return;
    int e = (int)(idx >> 2), h = (int)(idx & 3);
    int s = ei[e];
    int d = ei[EDGES + e];
    const float4* kv = (const float4*)(k_rel + (long)s * HID + h * HD);
    const float4* qv = (const float4*)(q     + (long)d * HID + h * HD);
    float acc = 0.f;
#pragma unroll
    for (int i = 0; i < 8; ++i) {
        float4 ka = kv[i], qa = qv[i];
        acc += ka.x * qa.x + ka.y * qa.y + ka.z * qa.z + ka.w * qa.w;
    }
    float lg = acc * p_rel[h] * 0.17677669529663688110f;  // 1/sqrt(32)
    logits[(long)e * NH + h] = lg;
    atomicMax(&node_max[(long)d * NH + h], f2ord(lg));
}

// ---------------------------------------------------------------------------
// Pass 2: e = exp(logit - max); accumulate per-(dst,head) sum.
// ---------------------------------------------------------------------------
__global__ void edge_exp_kernel(float* __restrict__ logits,
                                const int* __restrict__ ei,
                                const unsigned* __restrict__ node_max,
                                float* __restrict__ node_sum) {
    long idx = (long)blockIdx.x * blockDim.x + threadIdx.x;
    if (idx >= (long)EDGES * NH) return;
    int e = (int)(idx >> 2), h = (int)(idx & 3);
    int d = ei[EDGES + e];
    float mx = ord2f(node_max[(long)d * NH + h]);
    float ex = __expf(logits[(long)e * NH + h] - mx);
    logits[(long)e * NH + h] = ex;
    atomicAdd(&node_sum[(long)d * NH + h], ex);
}

// ---------------------------------------------------------------------------
// Pass 3: out[d, c] += (e / (sum + eps)) * v_rel[s, c]; one lane per element.
// ---------------------------------------------------------------------------
__global__ void edge_scatter_kernel(const float* __restrict__ logits,
                                    const int* __restrict__ ei,
                                    const float* __restrict__ node_sum,
                                    const float* __restrict__ v_rel,
                                    float* __restrict__ out) {
    long idx = (long)blockIdx.x * blockDim.x + threadIdx.x;
    if (idx >= (long)EDGES * HID) return;
    long e = idx >> 7;
    int  c = (int)(idx & 127);
    int  h = c >> 5;
    int s = ei[e];
    int d = ei[EDGES + e];
    float w = logits[e * NH + h] / (node_sum[(long)d * NH + h] + 1e-16f);
    atomicAdd(&out[(long)d * HID + c], w * v_rel[(long)s * HID + c]);
}

// ---------------------------------------------------------------------------
extern "C" void kernel_launch(void* const* d_in, const int* in_sizes, int n_in,
                              void* d_out, int out_size, void* d_ws, size_t ws_size,
                              hipStream_t stream) {
    const float* x_user    = (const float*)d_in[0];
    const float* x_item    = (const float*)d_in[1];
    // d_in[2] = edge_ui : unused (only feeds out["item"], never returned)
    const int*   edge_iu   = (const int*)d_in[3];
    const float* W_in_user = (const float*)d_in[4];
    const float* b_in_user = (const float*)d_in[5];
    const float* W_in_item = (const float*)d_in[6];
    const float* b_in_item = (const float*)d_in[7];
    const float* Wk        = (const float*)d_in[8];
    const float* bk        = (const float*)d_in[9];
    const float* Wq        = (const float*)d_in[10];
    const float* bq        = (const float*)d_in[11];
    const float* Wv        = (const float*)d_in[12];
    const float* bv        = (const float*)d_in[13];
    const float* Wa        = (const float*)d_in[14];
    const float* ba        = (const float*)d_in[15];
    const float* a_rel     = (const float*)d_in[16];
    const float* m_rel     = (const float*)d_in[17];
    const float* p_rel     = (const float*)d_in[18];
    const float* skip      = (const float*)d_in[19];

    // workspace layout (256B aligned)
    char* ws = (char*)d_ws;
    size_t off = 0;
    auto walloc = [&](size_t bytes) -> void* {
        void* p = ws + off;
        off = (off + bytes + 255) & ~(size_t)255;
        return p;
    };
    const size_t NODE_F = (size_t)N_NODES * HID * sizeof(float);  // 51.2 MB
    float* h_user   = (float*)walloc(NODE_F);
    float* h_item   = (float*)walloc(NODE_F);   // reused as out_user later
    float* q_user   = (float*)walloc(NODE_F);
    float* v_rel    = (float*)walloc(NODE_F);
    float* logits   = (float*)walloc((size_t)EDGES * NH * sizeof(float));
    unsigned* nmax  = (unsigned*)walloc((size_t)N_NODES * NH * sizeof(unsigned));
    float* nsum     = (float*)walloc((size_t)N_NODES * NH * sizeof(float));
    float* Wk_eff   = (float*)walloc(128 * 128 * sizeof(float));
    float* bk_eff   = (float*)walloc(128 * sizeof(float));
    float* Wv_eff   = (float*)walloc(128 * 128 * sizeof(float));
    float* bv_eff   = (float*)walloc(128 * sizeof(float));
    float* k_rel    = (float*)d_out;   // dead before final GEMM overwrites d_out
    float* out_user = h_item;          // h_item dead before scatter begins

    // zero the atomic accumulators (every call: harness replays the graph)
    hipMemsetAsync(nmax, 0, (size_t)N_NODES * NH * sizeof(unsigned), stream);
    hipMemsetAsync(nsum, 0, (size_t)N_NODES * NH * sizeof(float), stream);

    // fold relation transforms into K/V weights (relation r=1, item->user)
    {
        int n = 129 * 128, blk = 256, g = (n + blk - 1) / blk;
        fold_rel_kernel<<<g, blk, 0, stream>>>(Wk + 128 * 128, bk + 128,
                                               a_rel + NH * HD * HD, Wk_eff, bk_eff);
        fold_rel_kernel<<<g, blk, 0, stream>>>(Wv + 128 * 128, bv + 128,
                                               m_rel + NH * HD * HD, Wv_eff, bv_eff);
    }

    const int gN = (N_NODES + 127) / 128;  // 782 blocks of 8 waves

    // input projections (+relu)
    wmma_gemm_kernel<0, 1, false><<<gN, 256, 0, stream>>>(
        x_user, W_in_user, b_in_user, h_user, N_NODES, 128, nullptr, nullptr);
    wmma_gemm_kernel<0, 1, false><<<gN, 256, 0, stream>>>(
        x_item, W_in_item, b_in_item, h_item, N_NODES, 64, nullptr, nullptr);

    // q (user, type 0); k_rel / v_rel (item, type 1, relation folded in)
    wmma_gemm_kernel<0, 0, false><<<gN, 256, 0, stream>>>(
        h_user, Wq, bq, q_user, N_NODES, 128, nullptr, nullptr);
    wmma_gemm_kernel<0, 0, false><<<gN, 256, 0, stream>>>(
        h_item, Wk_eff, bk_eff, k_rel, N_NODES, 128, nullptr, nullptr);
    wmma_gemm_kernel<0, 0, false><<<gN, 256, 0, stream>>>(
        h_item, Wv_eff, bv_eff, v_rel, N_NODES, 128, nullptr, nullptr);

    // segment softmax + weighted scatter over item->user edges
    {
        long n = (long)EDGES * NH;
        int g = (int)((n + 255) / 256);
        edge_logits_kernel<<<g, 256, 0, stream>>>(k_rel, q_user, edge_iu,
                                                  p_rel + NH, logits, nmax);
        edge_exp_kernel<<<g, 256, 0, stream>>>(logits, edge_iu, nmax, nsum);
    }
    // h_item is dead now -> out_user aliases it; must zero before scatter
    hipMemsetAsync(out_user, 0, NODE_F, stream);
    {
        long n = (long)EDGES * HID;
        int g = (int)((n + 255) / 256);
        edge_scatter_kernel<<<g, 256, 0, stream>>>(logits, edge_iu, nsum,
                                                   v_rel, out_user);
    }

    // final: d_out = sig(skip0)*(gelu(out_user)@Wa0 + ba0) + (1-sig)*h_user
    wmma_gemm_kernel<1, 0, true><<<gN, 256, 0, stream>>>(
        out_user, Wa, ba, (float*)d_out, N_NODES, 128, h_user, skip);
}
